// CausalTemporalAttention_29618094474199
// MI455X (gfx1250) — compile-verified, matrix operand order checked by reading, not run on hardware
//
#include <hip/hip_runtime.h>
#include <hip/hip_bf16.h>

#define B_  8
#define W_  16
#define P_  8192
#define C_  64
#define G_  4
#define PT  64        // positions per workgroup tile
#define BLOCK 256     // 8 wave32

typedef __attribute__((ext_vector_type(16))) _Float16 v16h;
typedef __attribute__((ext_vector_type(2)))  _Float16 v2h;
typedef __attribute__((ext_vector_type(8)))  float    v8f;

// ---- fragment swizzle maps (CDNA5 ISA 7.12.2, 16-bit 16x32 fragments) ----
// element idx of lane 16h+n holds K = idx + 8h + (idx>=8 ? 8 : 0)

// B-operand (features): [kb2=c>>5][nT=p>>4][lane=16h+n][idx]
__device__ __forceinline__ int swz(int c, int p) {
    const int kk  = c & 31;
    const int h   = (kk >> 3) & 1;
    const int idx = (kk & 7) + ((kk & 16) ? 8 : 0);
    return ((((c >> 5) * 4 + (p >> 4)) * 32) + (h * 16 + (p & 15))) * 16 + idx;
}

// A-operand (weights, row o x col c): [mT=o>>4][kb2=c>>5][lane=16h+(o&15)][idx]
__device__ __forceinline__ int swzA(int o, int c) {
    const int kk  = c & 31;
    const int h   = (kk >> 3) & 1;
    const int idx = (kk & 7) + ((kk & 16) ? 8 : 0);
    return ((((o >> 4) * 2 + (c >> 5)) * 32) + (h * 16 + (o & 15))) * 16 + idx;
}

// D (f32 16x16): [mT=c>>4][nT=p>>4][lane=16h+(p&15)][r=c&7], h=(c>>3)&1
// For fixed (mT,nT,lane), r=0..7 is contiguous -> v8f loads/stores.
__device__ __forceinline__ int dswz(int c, int p) {
    const int rr = c & 15;
    return (((c >> 4) * 4 + (p >> 4)) * 32 + ((rr >> 3) * 16 + (p & 15))) * 8 + (rr & 7);
}

// Cooperative: convert 64x64 f32 weight -> pre-swizzled f16 A-fragments in LDS.
__device__ __forceinline__ void stage_weight(const float* __restrict__ Wg,
                                             _Float16* __restrict__ sWm, int tid) {
    #pragma unroll
    for (int e = tid; e < (C_ * C_) / 2; e += BLOCK) {   // pairs (c even, c+1)
        const int o  = e >> 5;
        const int cc = (e & 31) * 2;
        v2h hv;
        hv[0] = (_Float16)Wg[o * C_ + cc];
        hv[1] = (_Float16)Wg[o * C_ + cc + 1];
        *(v2h*)(sWm + swzA(o, cc)) = hv;                 // swzA(o,cc+1)==swzA(o,cc)+1
    }
}

// Per-wave bias broadcast: bias value for row m = mT*16 + 8h + r.
__device__ __forceinline__ v8f load_bias(const float* __restrict__ bg, int mT, int lane) {
    const int h = lane >> 4;
    v8f bv;
    #pragma unroll
    for (int r = 0; r < 8; ++r) bv[r] = bg[mT * 16 + 8 * h + r];
    return bv;
}

// D(dswz f32 LDS) = A(swizzled f16 LDS) * F(swizzled f16 LDS) + bias.
__device__ __forceinline__ void gemm_sw(const _Float16* __restrict__ sWm,
                                        v8f bias0, v8f bias1,
                                        const _Float16* __restrict__ sFs,
                                        float* __restrict__ Dout,
                                        int lane, int wave)
{
    const int nT  = wave & 3;
    const int mT0 = wave >> 2;
    const int mT1 = mT0 + 2;

    const v16h a00 = *(const v16h*)(sWm + ((mT0 * 2 + 0) * 32 + lane) * 16);
    const v16h a01 = *(const v16h*)(sWm + ((mT0 * 2 + 1) * 32 + lane) * 16);
    const v16h a10 = *(const v16h*)(sWm + ((mT1 * 2 + 0) * 32 + lane) * 16);
    const v16h a11 = *(const v16h*)(sWm + ((mT1 * 2 + 1) * 32 + lane) * 16);
    const v16h b0  = *(const v16h*)(sFs + ((0 * 4 + nT) * 32 + lane) * 16);
    const v16h b1  = *(const v16h*)(sFs + ((1 * 4 + nT) * 32 + lane) * 16);

    v8f acc0 = {}, acc1 = {};
    acc0 = __builtin_amdgcn_wmma_f32_16x16x32_f16(false, a00, false, b0, (short)0, acc0, false, false);
    acc1 = __builtin_amdgcn_wmma_f32_16x16x32_f16(false, a10, false, b0, (short)0, acc1, false, false);
    acc0 = __builtin_amdgcn_wmma_f32_16x16x32_f16(false, a01, false, b1, (short)0, acc0, false, false);
    acc1 = __builtin_amdgcn_wmma_f32_16x16x32_f16(false, a11, false, b1, (short)0, acc1, false, false);

    *(v8f*)(Dout + ((mT0 * 4 + nT) * 32 + lane) * 8) = acc0 + bias0;
    *(v8f*)(Dout + ((mT1 * 4 + nT) * 32 + lane) * 8) = acc1 + bias1;
}

// K projection fused with score partial: never materializes K.
// s[p] += sum over this wave's 32 rows of Q[c,p]*(K[c,p]); Q read as two contiguous v8f.
__device__ __forceinline__ void gemm_k_scores(const _Float16* __restrict__ sWm,
                                              v8f bias0, v8f bias1,
                                              const _Float16* __restrict__ sFs,
                                              const float* __restrict__ sQ,
                                              float* __restrict__ sS,
                                              int lane, int wave)
{
    const int nT  = wave & 3;
    const int mT0 = wave >> 2;
    const int mT1 = mT0 + 2;

    const v16h a00 = *(const v16h*)(sWm + ((mT0 * 2 + 0) * 32 + lane) * 16);
    const v16h a01 = *(const v16h*)(sWm + ((mT0 * 2 + 1) * 32 + lane) * 16);
    const v16h a10 = *(const v16h*)(sWm + ((mT1 * 2 + 0) * 32 + lane) * 16);
    const v16h a11 = *(const v16h*)(sWm + ((mT1 * 2 + 1) * 32 + lane) * 16);
    const v16h b0  = *(const v16h*)(sFs + ((0 * 4 + nT) * 32 + lane) * 16);
    const v16h b1  = *(const v16h*)(sFs + ((1 * 4 + nT) * 32 + lane) * 16);

    v8f acc0 = {}, acc1 = {};
    acc0 = __builtin_amdgcn_wmma_f32_16x16x32_f16(false, a00, false, b0, (short)0, acc0, false, false);
    acc1 = __builtin_amdgcn_wmma_f32_16x16x32_f16(false, a10, false, b0, (short)0, acc1, false, false);
    acc0 = __builtin_amdgcn_wmma_f32_16x16x32_f16(false, a01, false, b1, (short)0, acc0, false, false);
    acc1 = __builtin_amdgcn_wmma_f32_16x16x32_f16(false, a11, false, b1, (short)0, acc1, false, false);

    const v8f q0 = *(const v8f*)(sQ + ((mT0 * 4 + nT) * 32 + lane) * 8);
    const v8f q1 = *(const v8f*)(sQ + ((mT1 * 4 + nT) * 32 + lane) * 8);
    float s = 0.f;
    #pragma unroll
    for (int r = 0; r < 8; ++r)
        s += q0[r] * (acc0[r] + bias0[r]) + q1[r] * (acc1[r] + bias1[r]);

    const int p = nT * 16 + (lane & 15);
    atomicAdd(&sS[p], s);
}

// conv1d(3-tap, zero pad) + exact GELU; writes swizzled f16 features (paired b32 stores).
__device__ __forceinline__ void features_w(const float* __restrict__ u,
                                           const float* __restrict__ emb_w,
                                           const float* __restrict__ emb_b,
                                           _Float16* __restrict__ sFs,
                                           int p0, int tid)
{
    const int p  = tid & (PT - 1);
    const int q  = tid >> 6;            // 0..3
    const int pg = p0 + p;
    const float um1 = (pg == 0)      ? 0.f : u[pg - 1];
    const float u0  = u[pg];
    const float up1 = (pg == P_ - 1) ? 0.f : u[pg + 1];
    #pragma unroll
    for (int k = 0; k < 8; ++k) {
        const int c = 2 * (q + 4 * k);  // even channels 0..62
        const float x0 = emb_w[c * 3 + 0] * um1 + emb_w[c * 3 + 1] * u0 +
                         emb_w[c * 3 + 2] * up1 + emb_b[c];
        const float x1 = emb_w[c * 3 + 3] * um1 + emb_w[c * 3 + 4] * u0 +
                         emb_w[c * 3 + 5] * up1 + emb_b[c + 1];
        v2h hv;
        hv[0] = (_Float16)(0.5f * x0 * (1.f + erff(x0 * 0.70710678118654752f)));
        hv[1] = (_Float16)(0.5f * x1 * (1.f + erff(x1 * 0.70710678118654752f)));
        *(v2h*)(sFs + swz(c, p)) = hv;  // swz(c+1,p)==swz(c,p)+1 for even c
    }
}

__global__ void __launch_bounds__(64)
zero_stats_kernel(float* stats) {
    stats[threadIdx.x] = 0.f;   // 2 * B_ * G_ = 64 floats
}

__global__ void __launch_bounds__(BLOCK)
attn_kernel(const float* __restrict__ u_history,
            const float* __restrict__ emb_w, const float* __restrict__ emb_b,
            const float* __restrict__ q_w,   const float* __restrict__ q_b,
            const float* __restrict__ k_w,   const float* __restrict__ k_b,
            const float* __restrict__ v_w,   const float* __restrict__ v_b,
            const float* __restrict__ o_w,   const float* __restrict__ o_b,
            float* __restrict__ out,         // pre-norm x, (B,C,P)
            float* __restrict__ stats)       // [0..31] sums, [32..63] sumsq per (b,g)
{
    __shared__ alignas(32) _Float16 sFs[C_ * PT];        // swizzled f16 B operand (8 KB)
    __shared__ alignas(32) _Float16 sW[2 * C_ * C_];     // 2 weight slots, swizzled (16 KB)
    __shared__ alignas(32) float    sQ[C_ * PT];         // Q, dswz layout (16 KB)
    __shared__ alignas(32) float    sProj[C_ * PT];      // V / O GEMM D, dswz (16 KB)
    __shared__ float sS[PT];                             // raw score accumulator
    __shared__ float sM[PT], sD[PT];                     // online softmax state
    __shared__ float sAlpha[PT], sE[PT];                 // rescale / weight broadcast
    __shared__ float sGsum[G_], sGsq[G_];

    _Float16* sW0 = sW;
    _Float16* sW1 = sW + C_ * C_;

    const int tid  = threadIdx.x;
    const int lane = tid & 31;
    const int wave = tid >> 5;
    const int b    = blockIdx.x >> 7;               // P_/PT = 128 tiles per batch
    const int p0   = (blockIdx.x & 127) * PT;
    const int mT0  = wave >> 2;
    const int mT1  = mT0 + 2;
    const int pmy  = tid & (PT - 1);                // this thread's position
    const int qmy  = tid >> 6;                      // channel phase 0..3

    if (tid < G_) { sGsum[tid] = 0.f; sGsq[tid] = 0.f; }
    if (tid < PT) { sM[tid] = -INFINITY; sD[tid] = 0.f; }

    // ---- last-window features + q_w staging + Q projection ----
    stage_weight(q_w, sW0, tid);
    features_w(u_history + ((size_t)(b * W_ + (W_ - 1))) * P_, emb_w, emb_b, sFs, p0, tid);
    __syncthreads();
    gemm_sw(sW0, load_bias(q_b, mT0, lane), load_bias(q_b, mT1, lane),
            sFs, sQ, lane, wave);
    __syncthreads();

    // ---- stage k_w/v_w (Q GEMM has consumed sW0) ----
    stage_weight(k_w, sW0, tid);
    stage_weight(v_w, sW1, tid);
    const v8f bk0 = load_bias(k_b, mT0, lane), bk1 = load_bias(k_b, mT1, lane);
    const v8f bv0 = load_bias(v_b, mT0, lane), bv1 = load_bias(v_b, mT1, lane);

    // ---- single pass over windows: K->scores, online softmax, V->context(regs) ----
    float ctx[16];
    #pragma unroll
    for (int k = 0; k < 16; ++k) ctx[k] = 0.f;

    for (int w = 0; w < W_; ++w) {
        features_w(u_history + ((size_t)(b * W_ + w)) * P_, emb_w, emb_b, sFs, p0, tid);
        if (tid < PT) sS[tid] = 0.f;
        __syncthreads();                                    // A (covers staging on w==0)

        gemm_k_scores(sW0, bk0, bk1, sFs, sQ, sS, lane, wave);
        __syncthreads();                                    // B

        if (tid < PT) {                                     // online softmax update
            const float s  = sS[tid] * 0.125f;              // 1/sqrt(C)
            const float mo = sM[tid];
            const float mn = fmaxf(mo, s);
            const float al = expf(mo - mn);                 // 0 when mo==-inf
            const float e  = expf(s - mn);
            sD[tid]     = sD[tid] * al + e;
            sM[tid]     = mn;
            sAlpha[tid] = al;
            sE[tid]     = e;
        }
        gemm_sw(sW1, bv0, bv1, sFs, sProj, lane, wave);     // V (same interval: no conflicts)
        __syncthreads();                                    // C

        {
            const float al = sAlpha[pmy];
            const float e  = sE[pmy];
            #pragma unroll
            for (int k = 0; k < 16; ++k) {
                const int c = qmy + 4 * k;
                ctx[k] = ctx[k] * al + e * sProj[dswz(c, pmy)];
            }
        }
        // next iteration's features/zero-sS share this interval (disjoint buffers)
    }

    // ---- capture last-window features (still in sFs from w==15), normalize context ----
    float last[16];
    {
        const float dinv = 1.f / sD[pmy];
        #pragma unroll
        for (int k = 0; k < 16; ++k) {
            const int c = qmy + 4 * k;
            last[k] = (float)sFs[swz(c, pmy)];
            sFs[swz(c, pmy)] = (_Float16)(ctx[k] * dinv);   // self-owned slots only
        }
    }
    stage_weight(o_w, sW0, tid);                            // k_w no longer needed
    __syncthreads();
    gemm_sw(sW0, load_bias(o_b, mT0, lane), load_bias(o_b, mT1, lane),
            sFs, sProj, lane, wave);
    __syncthreads();

    // ---- residual + store x + group statistics ----
    {
        float psum[G_] = {0.f, 0.f, 0.f, 0.f};
        float psq[G_]  = {0.f, 0.f, 0.f, 0.f};
        #pragma unroll
        for (int k = 0; k < 16; ++k) {
            const int c = qmy + 4 * k;
            const float x = sProj[dswz(c, pmy)] + last[k];
            out[((size_t)(b * C_ + c)) * P_ + p0 + pmy] = x;
            const int g = c >> 4;
            psum[g] += x;
            psq[g]  += x * x;
        }
        #pragma unroll
        for (int g = 0; g < G_; ++g) {
            atomicAdd(&sGsum[g], psum[g]);
            atomicAdd(&sGsq[g],  psq[g]);
        }
    }
    __syncthreads();
    if (tid < G_) {
        atomicAdd(&stats[b * G_ + tid],           sGsum[tid]);
        atomicAdd(&stats[B_ * G_ + b * G_ + tid], sGsq[tid]);
    }
}

__global__ void __launch_bounds__(BLOCK)
groupnorm_finalize(const float* __restrict__ stats,
                   const float* __restrict__ gamma,
                   const float* __restrict__ beta,
                   float* __restrict__ out)
{
    const size_t idx = (size_t)blockIdx.x * BLOCK + threadIdx.x;  // B*C*P
    const int b = (int)(idx / ((size_t)C_ * P_));
    const int c = (int)((idx / P_) % C_);
    const int g = c >> 4;
    const float N    = (float)((C_ / G_) * P_);
    const float s    = stats[b * G_ + g];
    const float sq   = stats[B_ * G_ + b * G_ + g];
    const float mean = s / N;
    const float var  = sq / N - mean * mean;
    const float x    = out[idx];
    out[idx] = (x - mean) * rsqrtf(var + 1e-5f) * gamma[c] + beta[c];
}

extern "C" void kernel_launch(void* const* d_in, const int* in_sizes, int n_in,
                              void* d_out, int out_size, void* d_ws, size_t ws_size,
                              hipStream_t stream) {
    const float* u_history = (const float*)d_in[0];
    const float* emb_w     = (const float*)d_in[1];
    const float* emb_b     = (const float*)d_in[2];
    const float* q_w       = (const float*)d_in[3];
    const float* q_b       = (const float*)d_in[4];
    const float* k_w       = (const float*)d_in[5];
    const float* k_b       = (const float*)d_in[6];
    const float* v_w       = (const float*)d_in[7];
    const float* v_b       = (const float*)d_in[8];
    const float* o_w       = (const float*)d_in[9];
    const float* o_b       = (const float*)d_in[10];
    const float* gn_gamma  = (const float*)d_in[11];
    const float* gn_beta   = (const float*)d_in[12];
    float* out   = (float*)d_out;
    float* stats = (float*)d_ws;   // 64 floats: sums then sumsq

    zero_stats_kernel<<<1, 64, 0, stream>>>(stats);
    attn_kernel<<<B_ * (P_ / PT), BLOCK, 0, stream>>>(
        u_history, emb_w, emb_b, q_w, q_b, k_w, k_b, v_w, v_b, o_w, o_b, out, stats);
    groupnorm_finalize<<<(B_ * C_ * P_) / BLOCK, BLOCK, 0, stream>>>(
        stats, gn_gamma, gn_beta, out);
}
